// CRF_18915035971994
// MI455X (gfx1250) — matrix operand order repaired and checked by reading, not compiled, not run on hardware
//
#include <hip/hip_runtime.h>

typedef __attribute__((ext_vector_type(16))) _Float16 v16h;
typedef __attribute__((ext_vector_type(8)))  _Float16 v8h;
typedef __attribute__((ext_vector_type(8)))  float    v8f;

#define T_LEN 512
#define BATCH 64
#define NTAG  256
#define START_TAG 254
#define END_TAG   255
#define NEG_INF (-10000.0f)

// LDS layout (dynamic shared):
#define SMEM_E_OFF  0                                  // 256*256 f16 = 131072 B
#define SMEM_A_OFF  (NTAG * NTAG * 2)                  // 64*256 f16 = 32768 B
#define SMEM_M0_OFF (SMEM_A_OFF + BATCH * NTAG * 2)    // 64 u32
#define SMEM_M1_OFF (SMEM_M0_OFF + BATCH * 4)          // 64 u32
#define SMEM_BYTES  (SMEM_M1_OFF + BATCH * 4)          // 164352 B  (< 320KB WGP LDS)

// Monotone float <-> uint encoding so LDS u32 atomicMax == float max
__device__ __forceinline__ unsigned encf(float f) {
  unsigned u = __float_as_uint(f);
  return (u & 0x80000000u) ? ~u : (u | 0x80000000u);
}
__device__ __forceinline__ float decf(unsigned e) {
  return __uint_as_float((e & 0x80000000u) ? (e ^ 0x80000000u) : ~e);
}

// A-fragment for v_wmma_f32_16x16x32_f16: lane(0-15)=row M, lanes 16-31 same rows.
// VGPR0-7 hold K = klo..klo+15 contiguous (klo = 0 or 16 by half).
__device__ __forceinline__ v16h load_afrag(const _Float16* ah, int arow, int kcol) {
  union { v16h v; v8h h[2]; } u;
  u.h[0] = *(const v8h*)(ah + arow * NTAG + kcol);
  u.h[1] = *(const v8h*)(ah + arow * NTAG + kcol + 8);
  return u.v;
}

__global__ void __launch_bounds__(256) crf_forward_kernel(
    const float* __restrict__ unary,    // (B,T,N)
    const float* __restrict__ trans,    // (1,N,N) : trans[j*N+i]
    const int*   __restrict__ lengths,  // (B,)
    float*       __restrict__ alpha_out // (B,N) f32 workspace
) {
  extern __shared__ char smem[];
  _Float16* Eh   = (_Float16*)(smem + SMEM_E_OFF);  // Eh[j*256+i] = exp(tr[j][i])
  _Float16* ah   = (_Float16*)(smem + SMEM_A_OFF);  // ah[b*256+i] = exp(alpha-amax)
  unsigned* amax0 = (unsigned*)(smem + SMEM_M0_OFF);
  unsigned* amax1 = (unsigned*)(smem + SMEM_M1_OFF);

  const int tid  = threadIdx.x;
  const int lane = tid & 31;
  const int wave = tid >> 5;           // 8 waves
  const int mb   = (wave >> 2) * 32;   // batch-row block  (2 blocks of 32)
  const int jb   = (wave & 3) * 64;    // tag-col block    (4 blocks of 64)
  const int l15  = lane & 15;
  const int hi   = lane >> 4;          // 0/1 half of wave
  const int klo  = hi ? 16 : 0;
  const unsigned ENC_NEG = encf(-1.0e30f);

  // ---- one-time setup ----
  for (int e = tid; e < NTAG * NTAG; e += 256)
    Eh[e] = (_Float16)__expf(trans[e]);                       // row j contiguous in i (=K)
  for (int e = tid; e < BATCH * NTAG; e += 256)
    ah[e] = (_Float16)(((e & (NTAG - 1)) == START_TAG) ? 1.0f : 0.0f);
  if (tid < BATCH) { amax0[tid] = encf(0.0f); amax1[tid] = ENC_NEG; }
  __syncthreads();

  // Persistent B-fragments (E): lane holds column j = jb+jt*16+l15, contiguous K run.
  v16h bfrag[4][8];
#pragma unroll
  for (int jt = 0; jt < 4; ++jt)
#pragma unroll
    for (int k = 0; k < 8; ++k)
      bfrag[jt][k] = *(const v16h*)(Eh + (jb + jt * 16 + l15) * NTAG + k * 32 + (hi ? 16 : 0));

  // alpha tile in registers: al[mt][jt][r] -> (row = mb+mt*16+8*hi+r, col = jb+jt*16+l15)
  float al[2][4][8];
#pragma unroll
  for (int mt = 0; mt < 2; ++mt)
#pragma unroll
    for (int jt = 0; jt < 4; ++jt)
#pragma unroll
      for (int r = 0; r < 8; ++r)
        al[mt][jt][r] = ((jb + jt * 16 + l15) == START_TAG) ? 0.0f : NEG_INF;

  int lenr[2][8];
  int uoff[2][8];
#pragma unroll
  for (int mt = 0; mt < 2; ++mt)
#pragma unroll
    for (int r = 0; r < 8; ++r) {
      int m = mb + mt * 16 + 8 * hi + r;
      lenr[mt][r] = lengths[m];
      uoff[mt][r] = m * (T_LEN * NTAG) + jb + l15;
    }

  // ---- serial time loop: each step is a 64x256x256 f16 WMMA GEMM ----
  for (int t = 0; t < T_LEN; ++t) {
    unsigned* amx_cur = (t & 1) ? amax1 : amax0;  // holds amax^(t)
    unsigned* amx_nxt = (t & 1) ? amax0 : amax1;  // pre-set to -inf; builds amax^(t+1)
    const int tb = t * NTAG;

    v8f acc[2][4];
    v8f z = {};
#pragma unroll
    for (int mt = 0; mt < 2; ++mt)
#pragma unroll
      for (int jt = 0; jt < 4; ++jt) acc[mt][jt] = z;

#pragma unroll
    for (int k = 0; k < 8; ++k) {
      v16h af0 = load_afrag(ah, mb + l15,      k * 32 + klo);
      v16h af1 = load_afrag(ah, mb + 16 + l15, k * 32 + klo);
#pragma unroll
      for (int jt = 0; jt < 4; ++jt) {
        acc[0][jt] = __builtin_amdgcn_wmma_f32_16x16x32_f16(
            false, af0, false, bfrag[jt][k], (short)0, acc[0][jt], false, false);
        acc[1][jt] = __builtin_amdgcn_wmma_f32_16x16x32_f16(
            false, af1, false, bfrag[jt][k], (short)0, acc[1][jt], false, false);
      }
    }

    // alpha' = u_t + amax + log(S), masked by t < length
    float amv[2][8];
#pragma unroll
    for (int mt = 0; mt < 2; ++mt)
#pragma unroll
      for (int r = 0; r < 8; ++r)
        amv[mt][r] = decf(amx_cur[mb + mt * 16 + 8 * hi + r]);

#pragma unroll
    for (int mt = 0; mt < 2; ++mt)
#pragma unroll
      for (int r = 0; r < 8; ++r) {
        const bool upd = t < lenr[mt][r];
#pragma unroll
        for (int jt = 0; jt < 4; ++jt) {
          float uu   = unary[uoff[mt][r] + tb + jt * 16];
          float cand = uu + amv[mt][r] + __logf(acc[mt][jt][r]);
          al[mt][jt][r] = upd ? cand : al[mt][jt][r];
        }
      }

    // per-row max across the wave's 64 cols, then LDS atomic max across waves
#pragma unroll
    for (int mt = 0; mt < 2; ++mt)
#pragma unroll
      for (int r = 0; r < 8; ++r) {
        float v = fmaxf(fmaxf(al[mt][0][r], al[mt][1][r]),
                        fmaxf(al[mt][2][r], al[mt][3][r]));
        v = fmaxf(v, __shfl_xor(v, 1));
        v = fmaxf(v, __shfl_xor(v, 2));
        v = fmaxf(v, __shfl_xor(v, 4));
        v = fmaxf(v, __shfl_xor(v, 8));
        if (l15 == 0)
          atomicMax(&amx_nxt[mb + mt * 16 + 8 * hi + r], encf(v));
      }

    __syncthreads();  // (A) maxes done; a_lds reads done

    if (tid < BATCH) amx_cur[tid] = ENC_NEG;  // recycle as next step's "next" buffer

    // a^(t+1)[b,i] = exp(alpha - amax) -> f16 into LDS
#pragma unroll
    for (int mt = 0; mt < 2; ++mt)
#pragma unroll
      for (int r = 0; r < 8; ++r) {
        float amn = decf(amx_nxt[mb + mt * 16 + 8 * hi + r]);
#pragma unroll
        for (int jt = 0; jt < 4; ++jt) {
          float p = __expf(al[mt][jt][r] - amn);
          ah[(mb + mt * 16 + 8 * hi + r) * NTAG + jb + jt * 16 + l15] = (_Float16)p;
        }
      }

    __syncthreads();  // (C) new A visible before next GEMM
  }

  // snapshot final alpha for the finalize kernel
#pragma unroll
  for (int mt = 0; mt < 2; ++mt)
#pragma unroll
    for (int r = 0; r < 8; ++r)
#pragma unroll
      for (int jt = 0; jt < 4; ++jt)
        alpha_out[(mb + mt * 16 + 8 * hi + r) * NTAG + jb + jt * 16 + l15] = al[mt][jt][r];
}

// fwd_score = logsumexp_j(alpha + tr[END,j]); gold = sum of path scores; out = fwd - gold
__global__ void __launch_bounds__(256) crf_finalize_kernel(
    const float* __restrict__ unary, const float* __restrict__ trans,
    const int* __restrict__ tags, const int* __restrict__ lengths,
    const float* __restrict__ alphaF, float* __restrict__ out)
{
  __shared__ float red[256];
  const int b = blockIdx.x;
  const int tid = threadIdx.x;

  float x = alphaF[b * NTAG + tid] + trans[END_TAG * NTAG + tid];
  red[tid] = x; __syncthreads();
  for (int s = 128; s > 0; s >>= 1) { if (tid < s) red[tid] = fmaxf(red[tid], red[tid + s]); __syncthreads(); }
  float M = red[0]; __syncthreads();
  red[tid] = __expf(x - M); __syncthreads();
  for (int s = 128; s > 0; s >>= 1) { if (tid < s) red[tid] += red[tid + s]; __syncthreads(); }
  float fwd = M + __logf(red[0]);
  __syncthreads();

  const int L = lengths[b];
  float g = 0.0f;
  for (int tt = tid; tt < T_LEN; tt += 256) {
    if (tt < L) {
      int cur  = tags[b * T_LEN + tt];
      int prev = tt ? tags[b * T_LEN + tt - 1] : START_TAG;
      g += trans[cur * NTAG + prev] + unary[(b * T_LEN + tt) * NTAG + cur];
    }
  }
  red[tid] = g; __syncthreads();
  for (int s = 128; s > 0; s >>= 1) { if (tid < s) red[tid] += red[tid + s]; __syncthreads(); }

  if (tid == 0) {
    int last = tags[b * T_LEN + L - 1];
    out[b] = fwd - (red[0] + trans[END_TAG * NTAG + last]);
  }
}

extern "C" void kernel_launch(void* const* d_in, const int* in_sizes, int n_in,
                              void* d_out, int out_size, void* d_ws, size_t ws_size,
                              hipStream_t stream) {
  const float* unary   = (const float*)d_in[0];  // (B,T,N) f32
  const float* trans   = (const float*)d_in[1];  // (1,N,N) f32
  const int*   tags    = (const int*)d_in[2];    // (B,T) i32
  const int*   lengths = (const int*)d_in[3];    // (B,) i32
  float* out = (float*)d_out;                    // (B,) f32
  float* alphaWS = (float*)d_ws;                 // B*N f32 = 64KB scratch

  // allow >64KB dynamic LDS if the runtime gates it (no-op / ignored otherwise)
  (void)hipFuncSetAttribute((const void*)crf_forward_kernel,
                            hipFuncAttributeMaxDynamicSharedMemorySize, SMEM_BYTES);

  crf_forward_kernel<<<1, 256, SMEM_BYTES, stream>>>(unary, trans, lengths, alphaWS);
  crf_finalize_kernel<<<BATCH, 256, 0, stream>>>(unary, trans, tags, lengths, alphaWS, out);
}
